// ProteinGNN_8323646619820
// MI455X (gfx1250) — matrix operand order repaired and checked by reading, not compiled, not run on hardware
//
#include <hip/hip_runtime.h>
#include <hip/hip_bf16.h>

// ---------------------------------------------------------------------------
// Types for CDNA5 WMMA (wave32): D(16x16 f32) = A(16x32 bf16) x B(32x16 bf16) + C
// ---------------------------------------------------------------------------
typedef __attribute__((ext_vector_type(16))) __bf16 v16bf;
typedef __attribute__((ext_vector_type(8)))  float  v8f;

#define LN_EPS 1e-5f
#define MSG_WAVES 4   // waves per message workgroup (cooperative pw staging)
#define MSG_TPW   4   // edge tiles per wave

__device__ __forceinline__ unsigned short f2bfu(float f) {
    unsigned u = __builtin_bit_cast(unsigned, f);
    u += 0x7FFFu + ((u >> 16) & 1u);               // round-to-nearest-even
    return (unsigned short)(u >> 16);
}
__device__ __forceinline__ __bf16 f2bf(float f) {
    unsigned short s = f2bfu(f);
    return __builtin_bit_cast(__bf16, s);
}

// ---------------------------------------------------------------------------
// Utility kernels
// ---------------------------------------------------------------------------
__global__ void zero_kernel(float* p, long n) {
    long i = (long)blockIdx.x * blockDim.x + threadIdx.x;
    long stride = (long)gridDim.x * blockDim.x;
    for (; i < n; i += stride) p[i] = 0.0f;
}

// Pack a fp32 row-major [K x Ncols] weight into bf16 WMMA B-fragments.
// Fragment f=(kt*NT+nt); lane L holds col n = nt*16 + (L&15),
// h=L>>4; element j of the lane's v16bf is K = kt*32 + (j<8 ? 8h+j : 16+8h+(j-8)).
// Zero-pads k>=K and n>=Ncols.
__global__ void pack_b_bf16(const float* __restrict__ src, int K, int Ncols,
                            unsigned short* __restrict__ dst, int KT, int NT) {
    int t = blockIdx.x * blockDim.x + threadIdx.x;
    int total = KT * NT * 512;
    if (t >= total) return;
    int j    = t & 15;
    int lane = (t >> 4) & 31;
    int f    = t >> 9;
    int kt = f / NT, nt = f % NT;
    int k = kt * 32 + ((j & 8) << 1) + ((lane >> 4) << 3) + (j & 7);
    int n = nt * 16 + (lane & 15);
    float v = (k < K && n < Ncols) ? src[(size_t)k * Ncols + n] : 0.0f;
    dst[t] = f2bfu(v);
}

// ---------------------------------------------------------------------------
// Encoder: geometric invariants -> 16 -> 32, concat with x -> h0 [N x 64]
// (cols 56..63 zero-padded so layer-1 K dim is a multiple of 32)
// ---------------------------------------------------------------------------
__global__ void encoder_kernel(const float* __restrict__ x,
                               const float* __restrict__ pos,
                               const float* __restrict__ w1, const float* __restrict__ b1,
                               const float* __restrict__ w2, const float* __restrict__ b2,
                               float* __restrict__ h0, int N) {
    int n = blockIdx.x * blockDim.x + threadIdx.x;
    if (n >= N) return;
    float px = pos[n * 3 + 0], py = pos[n * 3 + 1], pz = pos[n * 3 + 2];
    float i0 = sqrtf(px * px + py * py + pz * pz);
    float i1 = px * px + py * py;
    float i2 = px * py;
    float hid[16];
#pragma unroll
    for (int j = 0; j < 16; ++j) {
        float t = i0 * w1[j] + i1 * w1[16 + j] + i2 * w1[32 + j] + b1[j];
        hid[j] = fmaxf(t, 0.0f);
    }
    float* o = h0 + (size_t)n * 64;
#pragma unroll
    for (int c = 0; c < 24; ++c) o[c] = x[(size_t)n * 24 + c];
    for (int c = 0; c < 32; ++c) {
        float g = b2[c];
#pragma unroll
        for (int j = 0; j < 16; ++j) g += hid[j] * w2[j * 32 + c];
        o[24 + c] = g;
    }
#pragma unroll
    for (int c = 56; c < 64; ++c) o[c] = 0.0f;
}

// ---------------------------------------------------------------------------
// Per-edge hidden: relu(LayerNorm(edge_attr @ ew1 + eb1)) -> bf16 [E x 16]
// Row-major so a lane's WMMA-A half-row is one 16-byte load.
// ---------------------------------------------------------------------------
__global__ void edge_hidden_kernel(const float* __restrict__ ea,
                                   const float* __restrict__ ew1, const float* __restrict__ eb1,
                                   const float* __restrict__ lng, const float* __restrict__ lnb,
                                   unsigned short* __restrict__ hiddenA, int E) {
    int e = blockIdx.x * blockDim.x + threadIdx.x;
    if (e >= E) return;
    float a[5];
#pragma unroll
    for (int i = 0; i < 5; ++i) a[i] = ea[(size_t)e * 5 + i];
    float t[16], s = 0.0f, q = 0.0f;
#pragma unroll
    for (int j = 0; j < 16; ++j) {
        float v = eb1[j];
#pragma unroll
        for (int i = 0; i < 5; ++i) v += a[i] * ew1[i * 16 + j];
        t[j] = v; s += v; q += v * v;
    }
    float mean = s * (1.0f / 16.0f);
    float var  = q * (1.0f / 16.0f) - mean * mean;   // biased, matches jnp.var
    float rstd = rsqrtf(var + LN_EPS);
#pragma unroll
    for (int j = 0; j < 16; ++j) {
        float v = fmaxf((t[j] - mean) * rstd * lng[j] + lnb[j], 0.0f);
        hiddenA[(size_t)e * 16 + j] = f2bfu(v);
    }
}

__global__ void edge_count_kernel(const int* __restrict__ dst, float* __restrict__ cnt, int E) {
    int e = blockIdx.x * blockDim.x + threadIdx.x;
    if (e >= E) return;
    unsafeAtomicAdd(&cnt[dst[e]], 1.0f);
}

// ---------------------------------------------------------------------------
// Message kernel: MSG_WAVES waves per block, MSG_TPW 16-edge tiles per wave.
//   Block: cooperatively stage packed pw fragments into LDS (once).
//   Per tile: edge_w = WMMA(hidden16 padK32, ew2) -> LDS transpose (wave-local,
//             LDS ops are in-order per wave -> wave_barrier suffices);
//             A fragments = bf16(x[src] * edge_w), resident for all kt;
//             msg tile = sum_kt WMMA(A[kt], pw[kt][nt] from LDS) -> f32 atomics.
//   x[src] rows prefetched (global_prefetch) as soon as src is known.
// ---------------------------------------------------------------------------
template <int DINP, int DINREAL, int DOUT>
__global__ __launch_bounds__(MSG_WAVES * 32)
void msg_kernel(const unsigned short* __restrict__ hiddenA,
                const float* __restrict__ hin,
                const unsigned short* __restrict__ ew2p,
                const float* __restrict__ eb2,
                const unsigned short* __restrict__ pwp,
                const int* __restrict__ src, const int* __restrict__ dst,
                float* __restrict__ aggr, int E) {
    constexpr int KT = DINP / 32;
    constexpr int NT = DOUT / 16;
    constexpr int PWB = KT * NT * 1024;     // bytes of packed pw fragments

    extern __shared__ char smem[];
    unsigned short* spw = (unsigned short*)smem;
    float* ew_base = (float*)(smem + PWB);

    const int tid  = threadIdx.x;
    const int wave = tid >> 5;
    const int lane = tid & 31;
    const int m16  = lane & 15;
    const int half = lane >> 4;

    float* ew_tile = ew_base + wave * (16 * 36);
    int*   sdst    = (int*)(ew_base + MSG_WAVES * (16 * 36)) + wave * 16;

    // ---- cooperative staging of pw fragments into LDS (16-byte chunks) ----
    {
        const uint4* g = (const uint4*)pwp;
        uint4* l = (uint4*)spw;
        for (int i = tid; i < PWB / 16; i += MSG_WAVES * 32) l[i] = g[i];
    }
    __syncthreads();

    const v16bf* ew2f = (const v16bf*)ew2p;
    const v16bf* spwf = (const v16bf*)spw;

    for (int t = 0; t < MSG_TPW; ++t) {
        const int tile = (blockIdx.x * MSG_WAVES + wave) * MSG_TPW + t;
        if (tile * 16 >= E) break;              // wave-uniform: EXEC stays full

        int e  = tile * 16 + m16;
        int ec = (e < E) ? e : (E - 1);
        if (half == 0) sdst[m16] = dst[ec];
        const int srow = src[ec];

        // Prefetch the gather row while we do the edge-MLP WMMAs.
        const float* xbase = hin + (size_t)srow * DINP;
        __builtin_prefetch(xbase, 0, 3);
        if (DINP > 32) __builtin_prefetch(xbase + DINP - 32, 0, 3);

        // WMMA A fragment of hidden (K padded 16->32 with zeros)
        v16bf Ah;
        {
            union { uint4 u; unsigned short s[8]; } th;
            th.u = *(const uint4*)(hiddenA + (size_t)ec * 16 + half * 8);
#pragma unroll
            for (int j = 0; j < 8; ++j) Ah[j] = __builtin_bit_cast(__bf16, th.s[j]);
#pragma unroll
            for (int j = 8; j < 16; ++j) Ah[j] = __builtin_bit_cast(__bf16, (unsigned short)0);
        }

        v16bf Am[KT];
#pragma unroll
        for (int kt = 0; kt < KT; ++kt) {
            // edge_w[:, kt*32 .. kt*32+31] in two 16-col WMMAs
#pragma unroll
            for (int n2 = 0; n2 < 2; ++n2) {
                v16bf B = ew2f[(kt * 2 + n2) * 32 + lane];
                v8f C = {};
                C = __builtin_amdgcn_wmma_f32_16x16x32_bf16(false, Ah, false, B,
                                                            (short)0, C, false, false);
                int col = kt * 32 + n2 * 16 + m16;
                float bias = (col < DINREAL) ? eb2[col] : 0.0f;
#pragma unroll
                for (int r = 0; r < 8; ++r)
                    ew_tile[(r + 8 * half) * 36 + n2 * 16 + m16] = C[r] + bias;
            }
            __builtin_amdgcn_wave_barrier();    // LDS in-order per wave: order only
            // Build A_msg fragment: elementwise x[src] * edge_w -> bf16
            const float4* x4 = (const float4*)(xbase + kt * 32 + half * 8);
            float4 xa = x4[0];           // K offsets +0..3
            float4 xb = x4[1];           // K offsets +4..7
            float4 xc = x4[4];           // K offsets +16..19
            float4 xd = x4[5];           // K offsets +20..23
            float xv[16] = {xa.x, xa.y, xa.z, xa.w, xb.x, xb.y, xb.z, xb.w,
                            xc.x, xc.y, xc.z, xc.w, xd.x, xd.y, xd.z, xd.w};
#pragma unroll
            for (int j = 0; j < 8; ++j)
                Am[kt][j] = f2bf(xv[j] * ew_tile[m16 * 36 + half * 8 + j]);
#pragma unroll
            for (int j = 0; j < 8; ++j)
                Am[kt][8 + j] = f2bf(xv[8 + j] * ew_tile[m16 * 36 + 16 + half * 8 + j]);
            __builtin_amdgcn_wave_barrier();
        }

        // Message GEMM over output tiles + scatter-add (pb added at node update)
        for (int nt = 0; nt < NT; ++nt) {
            v8f C = {};
#pragma unroll
            for (int kt = 0; kt < KT; ++kt) {
                v16bf B = spwf[(kt * NT + nt) * 32 + lane];   // ds_load_b128 x2
                C = __builtin_amdgcn_wmma_f32_16x16x32_bf16(false, Am[kt], false, B,
                                                            (short)0, C, false, false);
            }
            int col = nt * 16 + m16;
#pragma unroll
            for (int r = 0; r < 8; ++r) {
                int m  = r + 8 * half;
                int ee = tile * 16 + m;
                if (ee < E)
                    unsafeAtomicAdd(&aggr[(size_t)sdst[m] * DOUT + col], C[r]);
            }
        }
    }
}

// ---------------------------------------------------------------------------
// Node update: out = h @ nw + nb + (aggr + cnt*pb)/max(cnt,1)
// 1 wave per 16-node tile; waves own disjoint rows -> plain stores.
// ---------------------------------------------------------------------------
template <int DINP, int DOUT>
__global__ __launch_bounds__(32)
void node_kernel(const float* __restrict__ hin,
                 const unsigned short* __restrict__ nwp,
                 const float* __restrict__ nb, const float* __restrict__ pb,
                 const float* __restrict__ aggr, const float* __restrict__ cnt,
                 float* __restrict__ hout, int N) {
    constexpr int KT = DINP / 32;
    constexpr int NT = DOUT / 16;
    const int lane = threadIdx.x;
    const int m16  = lane & 15;
    const int half = lane >> 4;
    const int tile = blockIdx.x;

    int row = tile * 16 + m16;
    int rc  = (row < N) ? row : (N - 1);

    v16bf A[KT];
#pragma unroll
    for (int kt = 0; kt < KT; ++kt) {
        const float4* x4 = (const float4*)(hin + (size_t)rc * DINP + kt * 32 + half * 8);
        float4 xa = x4[0], xb = x4[1], xc = x4[4], xd = x4[5];
        float xv[16] = {xa.x, xa.y, xa.z, xa.w, xb.x, xb.y, xb.z, xb.w,
                        xc.x, xc.y, xc.z, xc.w, xd.x, xd.y, xd.z, xd.w};
#pragma unroll
        for (int j = 0; j < 16; ++j) A[kt][j] = f2bf(xv[j]);
    }

    const v16bf* nwf = (const v16bf*)nwp;
    for (int nt = 0; nt < NT; ++nt) {
        v8f C = {};
#pragma unroll
        for (int kt = 0; kt < KT; ++kt) {
            v16bf B = nwf[(kt * NT + nt) * 32 + lane];
            C = __builtin_amdgcn_wmma_f32_16x16x32_bf16(false, A[kt], false, B,
                                                        (short)0, C, false, false);
        }
        int col = nt * 16 + m16;
        float nbc = nb[col], pbc = pb[col];
#pragma unroll
        for (int r = 0; r < 8; ++r) {
            int rr = tile * 16 + r + 8 * half;
            if (rr < N) {
                float c = cnt[rr];
                float a = aggr[(size_t)rr * DOUT + col];
                float denom = fmaxf(c, 1.0f);
                hout[(size_t)rr * DOUT + col] = C[r] + nbc + (a + c * pbc) / denom;
            }
        }
    }
}

// ---------------------------------------------------------------------------
// BatchNorm (training mode, biased var) + ReLU, in place. C == 128.
// ---------------------------------------------------------------------------
__global__ void bn_stats_kernel(const float* __restrict__ h, float* __restrict__ sum,
                                float* __restrict__ sumsq, int N, int rowsPerBlock) {
    int c  = threadIdx.x;                 // 128 threads = 128 columns
    int r0 = blockIdx.x * rowsPerBlock;
    int r1 = r0 + rowsPerBlock; if (r1 > N) r1 = N;
    float s = 0.0f, q = 0.0f;
    for (int r = r0; r < r1; ++r) {
        float v = h[(size_t)r * 128 + c];
        s += v; q += v * v;
    }
    unsafeAtomicAdd(&sum[c], s);
    unsafeAtomicAdd(&sumsq[c], q);
}

__global__ void bn_apply_kernel(float* __restrict__ h, const float* __restrict__ sum,
                                const float* __restrict__ sumsq,
                                const float* __restrict__ g, const float* __restrict__ b,
                                int N) {
    long idx = (long)blockIdx.x * blockDim.x + threadIdx.x;
    long total = (long)N * 128;
    if (idx >= total) return;
    int c = (int)(idx & 127);
    float invN = 1.0f / (float)N;
    float m = sum[c] * invN;
    float v = sumsq[c] * invN - m * m;
    float val = (h[idx] - m) * rsqrtf(v + LN_EPS) * g[c] + b[c];
    h[idx] = fmaxf(val, 0.0f);
}

// ---------------------------------------------------------------------------
// Per-graph mean pooling
// ---------------------------------------------------------------------------
__global__ void pool_count_kernel(const int* __restrict__ batch, float* __restrict__ cg, int N) {
    int n = blockIdx.x * blockDim.x + threadIdx.x;
    if (n >= N) return;
    unsafeAtomicAdd(&cg[batch[n]], 1.0f);
}
__global__ void pool_accum_kernel(const float* __restrict__ h, const int* __restrict__ batch,
                                  float* __restrict__ pool, int N) {
    long idx = (long)blockIdx.x * blockDim.x + threadIdx.x;
    long total = (long)N * 256;
    if (idx >= total) return;
    int n = (int)(idx >> 8), c = (int)(idx & 255);
    unsafeAtomicAdd(&pool[(size_t)batch[n] * 256 + c], h[idx]);
}
__global__ void pool_final_kernel(const float* __restrict__ pool, const float* __restrict__ cg,
                                  float* __restrict__ out, int G) {
    int idx = blockIdx.x * blockDim.x + threadIdx.x;
    if (idx >= G * 256) return;
    out[idx] = pool[idx] / fmaxf(cg[idx >> 8], 1.0f);
}

// ---------------------------------------------------------------------------
// Host orchestration
// ---------------------------------------------------------------------------
extern "C" void kernel_launch(void* const* d_in, const int* in_sizes, int n_in,
                              void* d_out, int out_size, void* d_ws, size_t ws_size,
                              hipStream_t stream) {
    const int N = in_sizes[0] / 24;
    const int E = in_sizes[3] / 5;
    const int G = out_size / 256;

    const float* x      = (const float*)d_in[0];
    const float* pos    = (const float*)d_in[1];
    const int*   eidx   = (const int*)d_in[2];
    const float* eattr  = (const float*)d_in[3];
    const int*   batch  = (const int*)d_in[4];
    const float* enc_w1 = (const float*)d_in[6];
    const float* enc_b1 = (const float*)d_in[7];
    const float* enc_w2 = (const float*)d_in[8];
    const float* enc_b2 = (const float*)d_in[9];
    // layers: base 10/20/30; order: ew1, eb1, lng, lnb, ew2, eb2, pw, pb, nw, nb
    const float* L[3][10];
    for (int l = 0; l < 3; ++l)
        for (int k = 0; k < 10; ++k) L[l][k] = (const float*)d_in[10 + l * 10 + k];
    const float* bn1_g = (const float*)d_in[40];
    const float* bn1_b = (const float*)d_in[41];
    const float* bn2_g = (const float*)d_in[42];
    const float* bn2_b = (const float*)d_in[43];

    const int* src = eidx;
    const int* dst = eidx + E;

    // Workspace carve (working set ~190MB: L2-resident on MI455X)
    char* ws = (char*)d_ws;
    size_t off = 0;
    auto carve = [&](size_t bytes) -> void* {
        void* p = ws + off;
        off = (off + bytes + 255) & ~(size_t)255;
        return p;
    };
    float*          h0      = (float*)carve((size_t)N * 64 * 4);
    float*          h1      = (float*)carve((size_t)N * 128 * 4);
    float*          h2      = (float*)carve((size_t)N * 128 * 4);
    float*          h3      = (float*)carve((size_t)N * 256 * 4);
    unsigned short* hiddenA = (unsigned short*)carve((size_t)E * 16 * 2);
    float*          aggr    = (float*)carve((size_t)N * 256 * 4);
    float*          cnt     = (float*)carve((size_t)N * 4);
    unsigned short* ew2p[3], *pwp[3], *nwp[3];
    const int DINP[3] = {64, 128, 128};
    const int DINR[3] = {56, 128, 128};
    const int DOUT[3] = {128, 128, 256};
    for (int l = 0; l < 3; ++l) {
        int NTe = DINP[l] / 16;
        int KT = DINP[l] / 32, NT = DOUT[l] / 16;
        ew2p[l] = (unsigned short*)carve((size_t)1 * NTe * 512 * 2);
        pwp[l]  = (unsigned short*)carve((size_t)KT * NT * 512 * 2);
        nwp[l]  = (unsigned short*)carve((size_t)KT * NT * 512 * 2);
    }
    float* bnsum  = (float*)carve(128 * 4);
    float* bnsq   = (float*)carve(128 * 4);
    float* pool   = (float*)carve((size_t)G * 256 * 4);
    float* cntg   = (float*)carve((size_t)G * 4);

    const int TB = 256;
    auto blocks = [&](long n) { return (int)((n + TB - 1) / TB); };

    // Degree counts (identical for every layer)
    zero_kernel<<<256, TB, 0, stream>>>(cnt, N);
    edge_count_kernel<<<blocks(E), TB, 0, stream>>>(dst, cnt, E);

    // Encoder -> h0 [N x 64]
    encoder_kernel<<<blocks(N), TB, 0, stream>>>(x, pos, enc_w1, enc_b1, enc_w2, enc_b2, h0, N);

    // Pack weights into bf16 B-fragments
    for (int l = 0; l < 3; ++l) {
        int NTe = DINP[l] / 16;
        int KT = DINP[l] / 32, NT = DOUT[l] / 16;
        pack_b_bf16<<<blocks((long)1 * NTe * 512), TB, 0, stream>>>(
            L[l][4], 16, DINR[l], ew2p[l], 1, NTe);
        pack_b_bf16<<<blocks((long)KT * NT * 512), TB, 0, stream>>>(
            L[l][6], DINR[l], DOUT[l], pwp[l], KT, NT);
        pack_b_bf16<<<blocks((long)KT * NT * 512), TB, 0, stream>>>(
            L[l][8], DINR[l], DOUT[l], nwp[l], KT, NT);
    }

    const int etiles = (E + 15) / 16;
    const int ntiles = (N + 15) / 16;
    const int tilesPerBlock = MSG_WAVES * MSG_TPW;
    const int mblocks = (etiles + tilesPerBlock - 1) / tilesPerBlock;
    auto msgSmem = [&](int l) -> size_t {
        size_t pwb = (size_t)(DINP[l] / 32) * (DOUT[l] / 16) * 1024;
        return pwb + (size_t)MSG_WAVES * (16 * 36 * 4 + 16 * 4);
    };

    // ---- Layer 1: 64(pad of 56) -> 128, BN1 + ReLU ----
    edge_hidden_kernel<<<blocks(E), TB, 0, stream>>>(eattr, L[0][0], L[0][1], L[0][2], L[0][3], hiddenA, E);
    zero_kernel<<<512, TB, 0, stream>>>(aggr, (long)N * 128);
    msg_kernel<64, 56, 128><<<mblocks, MSG_WAVES * 32, msgSmem(0), stream>>>(
        hiddenA, h0, ew2p[0], L[0][5], pwp[0], src, dst, aggr, E);
    node_kernel<64, 128><<<ntiles, 32, 0, stream>>>(h0, nwp[0], L[0][9], L[0][7], aggr, cnt, h1, N);
    zero_kernel<<<1, TB, 0, stream>>>(bnsum, 128);
    zero_kernel<<<1, TB, 0, stream>>>(bnsq, 128);
    bn_stats_kernel<<<(N + 127) / 128, 128, 0, stream>>>(h1, bnsum, bnsq, N, 128);
    bn_apply_kernel<<<blocks((long)N * 128), TB, 0, stream>>>(h1, bnsum, bnsq, bn1_g, bn1_b, N);

    // ---- Layer 2: 128 -> 128, BN2 + ReLU ----
    edge_hidden_kernel<<<blocks(E), TB, 0, stream>>>(eattr, L[1][0], L[1][1], L[1][2], L[1][3], hiddenA, E);
    zero_kernel<<<512, TB, 0, stream>>>(aggr, (long)N * 128);
    msg_kernel<128, 128, 128><<<mblocks, MSG_WAVES * 32, msgSmem(1), stream>>>(
        hiddenA, h1, ew2p[1], L[1][5], pwp[1], src, dst, aggr, E);
    node_kernel<128, 128><<<ntiles, 32, 0, stream>>>(h1, nwp[1], L[1][9], L[1][7], aggr, cnt, h2, N);
    zero_kernel<<<1, TB, 0, stream>>>(bnsum, 128);
    zero_kernel<<<1, TB, 0, stream>>>(bnsq, 128);
    bn_stats_kernel<<<(N + 127) / 128, 128, 0, stream>>>(h2, bnsum, bnsq, N, 128);
    bn_apply_kernel<<<blocks((long)N * 128), TB, 0, stream>>>(h2, bnsum, bnsq, bn2_g, bn2_b, N);

    // ---- Layer 3: 128 -> 256 ----
    edge_hidden_kernel<<<blocks(E), TB, 0, stream>>>(eattr, L[2][0], L[2][1], L[2][2], L[2][3], hiddenA, E);
    zero_kernel<<<512, TB, 0, stream>>>(aggr, (long)N * 256);
    msg_kernel<128, 128, 256><<<mblocks, MSG_WAVES * 32, msgSmem(2), stream>>>(
        hiddenA, h2, ew2p[2], L[2][5], pwp[2], src, dst, aggr, E);
    node_kernel<128, 256><<<ntiles, 32, 0, stream>>>(h2, nwp[2], L[2][9], L[2][7], aggr, cnt, h3, N);

    // ---- Per-graph mean pooling -> d_out [G x 256] ----
    zero_kernel<<<64, TB, 0, stream>>>(pool, (long)G * 256);
    zero_kernel<<<1, TB, 0, stream>>>(cntg, G);
    pool_count_kernel<<<blocks(N), TB, 0, stream>>>(batch, cntg, N);
    pool_accum_kernel<<<blocks((long)N * 256), TB, 0, stream>>>(h3, batch, pool, N);
    pool_final_kernel<<<blocks((long)G * 256), TB, 0, stream>>>(pool, cntg, (float*)d_out, G);
}
